// ProbSparseAttention_36498632082001
// MI455X (gfx1250) — matrix operand order repaired and checked by reading, not compiled, not run on hardware
//
#include <hip/hip_runtime.h>
#include <hip/hip_bf16.h>
#include <math.h>

// ProbSparse attention for gfx1250 (MI455X), flash-style two-pass, bf16 WMMA.
// B=4 H=16 S=2048 D=64, fp32 in/out per the reference.
// Each wave owns 32 query rows (two A-fragments) so every B-fragment LDS load
// feeds two WMMAs. K^T/V B-fragments are gathered per element; P A-fragments are two 16B loads.

typedef __bf16 bf16_t;
typedef __attribute__((ext_vector_type(16))) __bf16 v16bf;  // 32 B
typedef __attribute__((ext_vector_type(8)))  __bf16 v8bf;   // 16 B
typedef __attribute__((ext_vector_type(4)))  __bf16 v4bf;   // 8 B
typedef __attribute__((ext_vector_type(8)))  float  v8f;

#define B_DIM   4
#define H_DIM   16
#define S_DIM   2048
#define D_DIM   64
#define QTILE   256       // query rows per workgroup (8 waves x 32)
#define MT      2         // 16-row M-tiles per wave
#define KTILE   32        // keys per inner iteration
#define NWAVES  8
#define SCALE   0.125f    // 1/sqrt(64)
#define SPARS   0.5f

// LDS row strides (elements). KT_STR/V_STR rows are 32B multiples so v16bf
// loads are naturally aligned; P_STR rows are 16B multiples for v8bf loads.
#define KT_STR  48        // 96 B
#define V_STR   80        // 160 B
#define P_STR   40        // 80 B

union PAunion { v16bf v; struct { v8bf lo, hi; } s; };

__global__ __launch_bounds__(256)
void probsparse_attn_kernel(const float* __restrict__ Q,
                            const float* __restrict__ K,
                            const float* __restrict__ V,
                            float* __restrict__ O) {
  // K stored TRANSPOSED: KldsT[dim][key]; QK^T B-fragment elements are read one dim row at a time.
  __shared__ __align__(32) bf16_t KldsT[D_DIM][KT_STR];            // 6144 B
  __shared__ __align__(32) bf16_t Vlds[KTILE][V_STR];              // 5120 B
  __shared__ __align__(16) bf16_t Plds[NWAVES][32][P_STR];         // 20480 B

  const int tid  = threadIdx.x;
  const int wave = tid >> 5;
  const int lane = tid & 31;
  const int lm   = lane & 15;
  const int hi   = lane >> 4;

  const int bh = blockIdx.y;                  // fused b*H + h
  const int q0 = blockIdx.x * QTILE + wave * 32;

  const float* __restrict__ Qb = Q + (size_t)bh * S_DIM * D_DIM;
  const float* __restrict__ Kb = K + (size_t)bh * S_DIM * D_DIM;
  const float* __restrict__ Vb = V + (size_t)bh * S_DIM * D_DIM;
  float* __restrict__       Ob = O + (size_t)bh * S_DIM * D_DIM;

  // ---- Q A-fragments (16x32 bf16): lane holds row m=lm of its M-tile,
  // dims [8*hi..8*hi+7] then [16+8*hi..+7] (A layout, ISA 7.12.2)
  v16bf a0[MT], a1[MT];
  #pragma unroll
  for (int mt = 0; mt < MT; ++mt) {
    const float* qrow = &Qb[(size_t)(q0 + mt * 16 + lm) * D_DIM];
    #pragma unroll
    for (int e = 0; e < 8; ++e) {
      a0[mt][e]     = (bf16_t)qrow[8 * hi + e];
      a0[mt][8 + e] = (bf16_t)qrow[16 + 8 * hi + e];
      a1[mt][e]     = (bf16_t)qrow[32 + 8 * hi + e];
      a1[mt][8 + e] = (bf16_t)qrow[48 + 8 * hi + e];
    }
  }

  // =========================== PASS 1: row stats ===========================
  float psum[MT][8], psq[MT][8], pmax[MT][8];
  #pragma unroll
  for (int mt = 0; mt < MT; ++mt)
    #pragma unroll
    for (int j = 0; j < 8; ++j) { psum[mt][j] = 0.f; psq[mt][j] = 0.f; pmax[mt][j] = -3.0e38f; }

  for (int kb = 0; kb < S_DIM; kb += KTILE) {
    __syncthreads();
    // stage K tile transposed: 32 keys x 64 dims fp32 -> bf16
    #pragma unroll
    for (int r = 0; r < 2; ++r) {
      const int idx = tid + r * 256;
      const int row = idx >> 4;               // key 0..31
      const int c4  = (idx & 15) * 4;         // dim 0..60
      const float4 f = *(const float4*)(&Kb[(size_t)(kb + row) * D_DIM + c4]);
      KldsT[c4 + 0][row] = (bf16_t)f.x;
      KldsT[c4 + 1][row] = (bf16_t)f.y;
      KldsT[c4 + 2][row] = (bf16_t)f.z;
      KldsT[c4 + 3][row] = (bf16_t)f.w;
    }
    if (kb + KTILE < S_DIM)
      __builtin_prefetch(&Kb[(size_t)(kb + KTILE) * D_DIM + tid * 8], 0, 1);
    __syncthreads();

    // issue ALL B-fragment loads up front (partial dscnt waits, DS/WMMA overlap)
    v16bf b0[2], b1[2];
    #pragma unroll
    for (int hh = 0; hh < 2; ++hh) {
      #pragma unroll
      for (int s = 0; s < 8; ++s) {
        b0[hh][s]     = KldsT[8 * hi + s][hh * 16 + lm];        // dims 0..31
        b0[hh][8 + s] = KldsT[16 + 8 * hi + s][hh * 16 + lm];
        b1[hh][s]     = KldsT[32 + 8 * hi + s][hh * 16 + lm];   // dims 32..63
        b1[hh][8 + s] = KldsT[48 + 8 * hi + s][hh * 16 + lm];
      }
    }
    #pragma unroll
    for (int hh = 0; hh < 2; ++hh) {
      #pragma unroll
      for (int mt = 0; mt < MT; ++mt) {
        v8f c = {};
        c = __builtin_amdgcn_wmma_f32_16x16x32_bf16(false, a1[mt], false, b1[hh], (short)0, c, false, false);
        c = __builtin_amdgcn_wmma_f32_16x16x32_bf16(false, a0[mt], false, b0[hh], (short)0, c, false, false);
        #pragma unroll
        for (int j = 0; j < 8; ++j) {
          const float s = c[j] * SCALE;
          psum[mt][j] += s;
          psq[mt][j]  += s * s;
          pmax[mt][j]  = fmaxf(pmax[mt][j], s);
        }
      }
    }
  }

  // reduce across the 16-lane half holding each row's columns
  float thr[MT][8], mrow[MT][8];
  #pragma unroll
  for (int mt = 0; mt < MT; ++mt) {
    #pragma unroll
    for (int j = 0; j < 8; ++j) {
      #pragma unroll
      for (int off = 1; off < 16; off <<= 1) {
        psum[mt][j] += __shfl_xor(psum[mt][j], off, 32);
        psq[mt][j]  += __shfl_xor(psq[mt][j],  off, 32);
        pmax[mt][j]  = fmaxf(pmax[mt][j], __shfl_xor(pmax[mt][j], off, 32));
      }
      const float mean = psum[mt][j] * (1.0f / (float)S_DIM);
      float var = (psq[mt][j] - (float)S_DIM * mean * mean) * (1.0f / (float)(S_DIM - 1));
      var = fmaxf(var, 0.0f);
      thr[mt][j]  = mean + SPARS * sqrtf(var);   // unbiased std (ddof=1)
      mrow[mt][j] = pmax[mt][j];
    }
  }

  // ====================== PASS 2: masked softmax + P*V =====================
  v8f oacc[MT][4];
  #pragma unroll
  for (int mt = 0; mt < MT; ++mt)
    #pragma unroll
    for (int nb = 0; nb < 4; ++nb) oacc[mt][nb] = (v8f){};
  float lsum[MT][8];
  #pragma unroll
  for (int mt = 0; mt < MT; ++mt)
    #pragma unroll
    for (int j = 0; j < 8; ++j) lsum[mt][j] = 0.f;

  for (int kb = 0; kb < S_DIM; kb += KTILE) {
    __syncthreads();
    // stage K (transposed) and V (row-major, packed b64 stores)
    #pragma unroll
    for (int r = 0; r < 2; ++r) {
      const int idx = tid + r * 256;
      const int row = idx >> 4;
      const int c4  = (idx & 15) * 4;
      const float4 fk = *(const float4*)(&Kb[(size_t)(kb + row) * D_DIM + c4]);
      KldsT[c4 + 0][row] = (bf16_t)fk.x;
      KldsT[c4 + 1][row] = (bf16_t)fk.y;
      KldsT[c4 + 2][row] = (bf16_t)fk.z;
      KldsT[c4 + 3][row] = (bf16_t)fk.w;
      const float4 fv = *(const float4*)(&Vb[(size_t)(kb + row) * D_DIM + c4]);
      const v4bf vv = { (bf16_t)fv.x, (bf16_t)fv.y, (bf16_t)fv.z, (bf16_t)fv.w };
      *(v4bf*)&Vlds[row][c4] = vv;
    }
    if (kb + KTILE < S_DIM) {
      __builtin_prefetch(&Kb[(size_t)(kb + KTILE) * D_DIM + tid * 8], 0, 1);
      __builtin_prefetch(&Vb[(size_t)(kb + KTILE) * D_DIM + tid * 8], 0, 1);
    }
    __syncthreads();

    // issue K^T and V fragment loads up front; V frags land before P scatter
    v16bf b0[2], b1[2], bv[4];
    #pragma unroll
    for (int hh = 0; hh < 2; ++hh) {
      #pragma unroll
      for (int s = 0; s < 8; ++s) {
        b0[hh][s]     = KldsT[8 * hi + s][hh * 16 + lm];
        b0[hh][8 + s] = KldsT[16 + 8 * hi + s][hh * 16 + lm];
        b1[hh][s]     = KldsT[32 + 8 * hi + s][hh * 16 + lm];
        b1[hh][8 + s] = KldsT[48 + 8 * hi + s][hh * 16 + lm];
      }
    }
    #pragma unroll
    for (int nb = 0; nb < 4; ++nb) {
      #pragma unroll
      for (int s = 0; s < 8; ++s) {
        bv[nb][s]     = Vlds[8 * hi + s][nb * 16 + lm];   // lane = dim col, elems = keys
        bv[nb][8 + s] = Vlds[16 + 8 * hi + s][nb * 16 + lm];
      }
    }

    // recompute scores, threshold-mask, exponentiate, round P to bf16 in LDS
    #pragma unroll
    for (int hh = 0; hh < 2; ++hh) {
      #pragma unroll
      for (int mt = 0; mt < MT; ++mt) {
        v8f c = {};
        c = __builtin_amdgcn_wmma_f32_16x16x32_bf16(false, a1[mt], false, b1[hh], (short)0, c, false, false);
        c = __builtin_amdgcn_wmma_f32_16x16x32_bf16(false, a0[mt], false, b0[hh], (short)0, c, false, false);
        #pragma unroll
        for (int j = 0; j < 8; ++j) {
          const float s = c[j] * SCALE;
          const float p = (s >= thr[mt][j]) ? __expf(s - mrow[mt][j]) : 0.0f;
          lsum[mt][j] += p;
          // C layout: row = j + 8*hi, col = lm (+16 for second half)
          Plds[wave][mt * 16 + j + 8 * hi][hh * 16 + lm] = (bf16_t)p;
        }
      }
    }
    // same-wave LDS store -> load (C-layout -> A-layout relayout)
    asm volatile("s_wait_dscnt 0x0" ::: "memory");

    // P A-fragments: two contiguous 16B halves per lane (A-layout order)
    #pragma unroll
    for (int mt = 0; mt < MT; ++mt) {
      PAunion pa;
      pa.s.lo = *(const v8bf*)&Plds[wave][mt * 16 + lm][8 * hi];
      pa.s.hi = *(const v8bf*)&Plds[wave][mt * 16 + lm][16 + 8 * hi];
      #pragma unroll
      for (int nb = 0; nb < 4; ++nb)
        oacc[mt][nb] = __builtin_amdgcn_wmma_f32_16x16x32_bf16(false, pa.v, false, bv[nb], (short)0,
                                                               oacc[mt][nb], false, false);
    }
  }

  // reduce softmax denominators and write O
  #pragma unroll
  for (int mt = 0; mt < MT; ++mt) {
    float inv[8];
    #pragma unroll
    for (int j = 0; j < 8; ++j) {
      #pragma unroll
      for (int off = 1; off < 16; off <<= 1)
        lsum[mt][j] += __shfl_xor(lsum[mt][j], off, 32);
      inv[j] = (lsum[mt][j] > 0.f) ? (1.0f / lsum[mt][j]) : 0.0f;
    }
    #pragma unroll
    for (int nb = 0; nb < 4; ++nb) {
      #pragma unroll
      for (int j = 0; j < 8; ++j) {
        const int row = q0 + mt * 16 + j + 8 * hi;
        const int col = nb * 16 + lm;
        Ob[(size_t)row * D_DIM + col] = oacc[mt][nb][j] * inv[j];
      }
    }
  }
}

extern "C" void kernel_launch(void* const* d_in, const int* in_sizes, int n_in,
                              void* d_out, int out_size, void* d_ws, size_t ws_size,
                              hipStream_t stream) {
  (void)in_sizes; (void)n_in; (void)out_size; (void)d_ws; (void)ws_size;
  const float* Q = (const float*)d_in[0];
  const float* K = (const float*)d_in[1];
  const float* V = (const float*)d_in[2];
  float* O = (float*)d_out;

  dim3 grid(S_DIM / QTILE, B_DIM * H_DIM);   // (8, 64)
  dim3 block(256);                            // 8 wave32
  probsparse_attn_kernel<<<grid, block, 0, stream>>>(Q, K, V, O);
}
